// EgoAttentionNetwork_6725918785920
// MI455X (gfx1250) — matrix operand order repaired and checked by reading, not compiled, not run on hardware
//
#include <hip/hip_runtime.h>

// ---------------------------------------------------------------------------
// EgoAttentionNetwork forward, MI455X (gfx1250), wave32 + WMMA f16->f32.
//
// Prologue kernel: swizzle oth_w1 / Wk / Wv into exact WMMA B-fragment order
// (f16) in d_ws once per call; 24KB stays L2-resident for all 8192 blocks.
// Main kernel: one workgroup (128 thr = 4 wave32) per batch element.
//  - 3x 64x64x64 GEMMs on v_wmma_f32_16x16x32_f16
//  - A fragments kept in LDS in fragment order  -> ds_load_b128
//  - B fragments loaded straight from global    -> global_load_b128 (L2 hit)
//  - K row-major / V transposed in LDS so attention reads are b128-wide
// ---------------------------------------------------------------------------

typedef __attribute__((ext_vector_type(16))) _Float16 v16h;
typedef __attribute__((ext_vector_type(8)))  float    v8f;

#define BT   128   // threads per block (4 wave32)
#define E_   64    // entities
#define D_   64    // feature size
#define FIN  7     // raw input features
#define NH   4     // heads
#define HD_  16    // per-head dim
#define EP   72    // padded f16 row stride (bank-conflict break; 144B = 9*16B)

union V16H { v16h v; uint4 q[2]; _Float16 h[16]; };
union PK2  { _Float16 h[2]; unsigned u; };

// ---- fragment-layout address helpers --------------------------------------
// B-fragment storage: wsw[(((mat*4 + nt)*2 + ks)*32 + lane)*16 + i]
//   lane = (d&15) + 16*(((k&31)>>3)&1), i = (k&7) + 8*((k&31)>>4)   (per ISA 8/16-bit B layout)
__device__ __forceinline__ int bFragOff(int mat, int nt, int ks, int lane) {
  return (((mat * 4 + nt) * 2 + ks) * 32 + lane) * 16;
}
// A-fragment storage: af[((strip*2 + ks)*32 + laneA)*16 + elem]
//   laneA = (m&15) + 16*(((k&31)>>3)&1), elem = (k&7) + 8*((k>>4)&1)
__device__ __forceinline__ int aOff(int m, int k) {
  const int strip = m >> 4, r = m & 15;
  const int ks = k >> 5, kk = k & 31;
  const int half = (kk >> 3) & 1;
  const int elem = (kk & 7) + 8 * ((kk >> 4) & 1);
  return ((strip * 2 + ks) * 32 + (r + 16 * half)) * 16 + elem;
}

__device__ __forceinline__ v16h load_afrag_lds(const _Float16* af, int strip,
                                               int ks, int lane) {
  const uint4* p =
      reinterpret_cast<const uint4*>(af + ((strip * 2 + ks) * 32 + lane) * 16);
  V16H r;
  r.q[0] = p[0];
  r.q[1] = p[1];
  return r.v;
}

__device__ __forceinline__ v16h load_bfrag_g(const _Float16* wsw, int mat,
                                             int nt, int ks, int lane) {
  const uint4* p = reinterpret_cast<const uint4*>(wsw + bFragOff(mat, nt, ks, lane));
  V16H r;
  r.q[0] = p[0];
  r.q[1] = p[1];
  return r.v;
}

// ---------------------------------------------------------------------------
// Prologue: convert + swizzle the three GEMM weight matrices into B-fragment
// order. 6144 (k-pair, d) elements -> one packed b32 store each.
// ---------------------------------------------------------------------------
__global__ __launch_bounds__(256) void stage_weights_kernel(
    const float* __restrict__ oth_w1, const float* __restrict__ Wk,
    const float* __restrict__ Wv, _Float16* __restrict__ wsw) {
  const int p = blockIdx.x * 256 + threadIdx.x;  // 0 .. 6143
  if (p >= 3 * 2048) return;
  const int mat = p >> 11;
  const int rem = p & 2047;
  const int d = rem & 63;
  const int k = (rem >> 6) * 2;  // even k: (k,k+1) contiguous in fragment
  const float* src = (mat == 0) ? oth_w1 : ((mat == 1) ? Wk : Wv);
  const float v0 = src[k * 64 + d];
  const float v1 = src[(k + 1) * 64 + d];
  const int kk = k & 31, ks = k >> 5;
  const int lane = (d & 15) + 16 * ((kk >> 3) & 1);
  const int i = (kk & 7) + 8 * (kk >> 4);
  PK2 pk;
  pk.h[0] = (_Float16)v0;
  pk.h[1] = (_Float16)v1;
  *reinterpret_cast<unsigned*>(wsw + bFragOff(mat, d >> 4, ks, lane) + i) = pk.u;
}

// ---------------------------------------------------------------------------
struct SMem {
  union {
    _Float16 hf[4096];       // layer0 output, A-fragment order
    _Float16 Ks[E_][EP];     // K, row-major (written in stage D)
  };
  _Float16 actf[4096];       // input_all, A-fragment order
  _Float16 Vt[D_][EP];       // V transposed: Vt[d][e]
  float xs[E_][8];           // raw x tile
  float w0s[2][FIN][D_];     // layer0 weights: [0]=ego, [1]=oth
  float b0s[2][D_];
  float b1o[D_];             // oth_b1
  float ego[D_];             // ego embedding (fp32)
  float qs[D_];              // q * 1/sqrt(HD)
  float scores[NH][E_];
  float p[NH][E_];
  float sred[NH];
  float outs[D_];
  float maskv[E_];
};

__global__ __launch_bounds__(BT) void ego_attention_kernel(
    const float* __restrict__ x,
    const float* __restrict__ ego_w0, const float* __restrict__ ego_b0,
    const float* __restrict__ ego_w1, const float* __restrict__ ego_b1,
    const float* __restrict__ oth_w0, const float* __restrict__ oth_b0,
    const float* __restrict__ oth_b1,
    const float* __restrict__ Wq, const float* __restrict__ Wc,
    const _Float16* __restrict__ wsw,   // swizzled B fragments (L2-resident)
    float* __restrict__ out) {
  __shared__ SMem sm;
  const int t    = threadIdx.x;
  const int lane = t & 31;
  const int wave = t >> 5;
  const int b    = blockIdx.x;

  // ---- Stage A: x tile + layer0 weights/biases into LDS (fp32) ------------
  const float* xb = x + (size_t)b * (E_ * FIN);
  for (int i = t; i < E_ * FIN; i += BT) {
    sm.xs[i / FIN][i % FIN] = xb[i];
    sm.w0s[0][i / 64][i & 63] = ego_w0[i];
    sm.w0s[1][i / 64][i & 63] = oth_w0[i];
  }
  if (t < D_) {
    sm.b0s[0][t] = ego_b0[t];
    sm.b0s[1][t] = oth_b0[t];
    sm.b1o[t]    = oth_b1[t];
  }
  __syncthreads();

  // ---- Stage B: mask + layer0 (K=7 VALU), write h in A-fragment order -----
  if (t < E_) sm.maskv[t] = (sm.xs[t][0] < 0.5f) ? 1.f : 0.f;
  for (int pp = t; pp < E_ * D_ / 2; pp += BT) {
    const int e = pp >> 5;            // 32 d-pairs per entity
    const int d0 = (pp & 31) * 2;
    const int sel = (e == 0) ? 0 : 1;
    float a0 = sm.b0s[sel][d0];
    float a1 = sm.b0s[sel][d0 + 1];
#pragma unroll
    for (int f = 0; f < FIN; ++f) {
      const float xv = sm.xs[e][f];
      a0 += xv * sm.w0s[sel][f][d0];
      a1 += xv * sm.w0s[sel][f][d0 + 1];
    }
    PK2 pk;
    pk.h[0] = (_Float16)fmaxf(a0, 0.f);
    pk.h[1] = (_Float16)fmaxf(a1, 0.f);
    *reinterpret_cast<unsigned*>(&sm.hf[aOff(e, d0)]) = pk.u;  // b32 store
  }
  __syncthreads();

  // ---- Stage C: layer1 (others weights), WMMA; D -> act in A-frag order ---
  {
    const v16h a0 = load_afrag_lds(sm.hf, wave, 0, lane);
    const v16h a1 = load_afrag_lds(sm.hf, wave, 1, lane);
    const int c = lane & 15, half = lane >> 4;
    const int halfA = (c >> 3) & 1;
#pragma unroll
    for (int nt = 0; nt < 4; ++nt) {
      const v16h b0 = load_bfrag_g(wsw, 0, nt, 0, lane);
      const v16h b1 = load_bfrag_g(wsw, 0, nt, 1, lane);
      v8f acc = {};
      acc = __builtin_amdgcn_wmma_f32_16x16x32_f16(false, a0, false, b0,
                                                   (short)0, acc, false, false);
      acc = __builtin_amdgcn_wmma_f32_16x16x32_f16(false, a1, false, b1,
                                                   (short)0, acc, false, false);
      const float bias = sm.b1o[nt * 16 + c];
      const int elem = (c & 7) + 8 * (nt & 1);
      const int ksA = nt >> 1;
#pragma unroll
      for (int j = 0; j < 8; ++j) {
        const int r = j + 8 * half;
        sm.actf[((wave * 2 + ksA) * 32 + (r + 16 * halfA)) * 16 + elem] =
            (_Float16)fmaxf(acc[j] + bias, 0.f);
      }
    }
  }
  __syncthreads();

  // ---- Stage C2: ego row uses ego_w1/ego_b1 — fp32 fixup of row 0 ---------
  if (t < D_) {
    float acc = ego_b1[t];
#pragma unroll 8
    for (int k = 0; k < D_; ++k)
      acc += (float)sm.hf[aOff(0, k)] * ego_w1[k * 64 + t];
    const float v = fmaxf(acc, 0.f);
    sm.ego[t] = v;
    sm.actf[aOff(0, t)] = (_Float16)v;
  }
  __syncthreads();

  // ---- Stage D: K and V projections (WMMA). K row-major, V transposed. ----
  // (Ks aliases hf, which is dead after the barrier above.)
  {
    const v16h a0 = load_afrag_lds(sm.actf, wave, 0, lane);
    const v16h a1 = load_afrag_lds(sm.actf, wave, 1, lane);
    const int c = lane & 15, half = lane >> 4;
    const int m0 = wave * 16;
#pragma unroll
    for (int nt = 0; nt < 4; ++nt) {
      const int n0 = nt * 16;
      const v16h bk0 = load_bfrag_g(wsw, 1, nt, 0, lane);
      const v16h bk1 = load_bfrag_g(wsw, 1, nt, 1, lane);
      v8f ka = {};
      ka = __builtin_amdgcn_wmma_f32_16x16x32_f16(false, a0, false, bk0,
                                                  (short)0, ka, false, false);
      ka = __builtin_amdgcn_wmma_f32_16x16x32_f16(false, a1, false, bk1,
                                                  (short)0, ka, false, false);
      const v16h bv0 = load_bfrag_g(wsw, 2, nt, 0, lane);
      const v16h bv1 = load_bfrag_g(wsw, 2, nt, 1, lane);
      v8f va = {};
      va = __builtin_amdgcn_wmma_f32_16x16x32_f16(false, a0, false, bv0,
                                                  (short)0, va, false, false);
      va = __builtin_amdgcn_wmma_f32_16x16x32_f16(false, a1, false, bv1,
                                                  (short)0, va, false, false);
#pragma unroll
      for (int j = 0; j < 8; ++j)
        sm.Ks[m0 + j + 8 * half][n0 + c] = (_Float16)ka[j];
#pragma unroll
      for (int j = 0; j < 8; j += 2) {   // V columns contiguous -> b32 stores
        PK2 pk;
        pk.h[0] = (_Float16)va[j];
        pk.h[1] = (_Float16)va[j + 1];
        *reinterpret_cast<unsigned*>(&sm.Vt[n0 + c][m0 + 8 * half + j]) = pk.u;
      }
    }
  }
  __syncthreads();

  // ---- Stage E: q = ego @ Wq, folded 1/sqrt(HD) ---------------------------
  if (t < D_) {
    float acc = 0.f;
#pragma unroll 8
    for (int k = 0; k < D_; ++k) acc += sm.ego[k] * Wq[k * 64 + t];
    sm.qs[t] = acc * 0.25f;
  }
  __syncthreads();

  // ---- Stage F1: masked scores; K row read = 2x ds_load_b128 --------------
  for (int i = t; i < NH * E_; i += BT) {
    const int hh = i >> 6, e = i & 63;
    V16H kr;
    const uint4* kp = reinterpret_cast<const uint4*>(&sm.Ks[e][hh * HD_]);
    kr.q[0] = kp[0];
    kr.q[1] = kp[1];
    float s = 0.f;
#pragma unroll
    for (int d = 0; d < HD_; ++d) s += sm.qs[hh * HD_ + d] * (float)kr.h[d];
    sm.scores[hh][e] = (sm.maskv[e] != 0.f) ? -1e9f : s;
  }
  __syncthreads();

  // ---- Stage F2: per-head softmax -----------------------------------------
  if (t < NH) {
    float m = -3.4e38f;
    for (int e = 0; e < E_; ++e) m = fmaxf(m, sm.scores[t][e]);
    float ssum = 0.f;
    for (int e = 0; e < E_; ++e) {
      const float ex = __expf(sm.scores[t][e] - m);
      sm.p[t][e] = ex;
      ssum += ex;
    }
    sm.sred[t] = 1.f / ssum;
  }
  __syncthreads();

  // ---- Stage F3: out[t] = sum_e p[h][e] * V[e][t]; V row = 8x b128 --------
  if (t < D_) {
    const int hh = t >> 4;
    union { uint4 q[8]; _Float16 h[64]; } vr;
    const uint4* vp = reinterpret_cast<const uint4*>(&sm.Vt[t][0]);
#pragma unroll
    for (int i = 0; i < 8; ++i) vr.q[i] = vp[i];
    float acc = 0.f;
#pragma unroll 8
    for (int e = 0; e < E_; ++e) acc += sm.p[hh][e] * (float)vr.h[e];
    sm.outs[t] = acc * sm.sred[hh];
  }
  __syncthreads();

  // ---- Stage G: result = (out @ Wc + ego) * 0.5 ---------------------------
  if (t < D_) {
    float acc = 0.f;
#pragma unroll 8
    for (int k = 0; k < D_; ++k) acc += sm.outs[k] * Wc[k * 64 + t];
    out[(size_t)b * D_ + t] = (acc + sm.ego[t]) * 0.5f;
  }
}

extern "C" void kernel_launch(void* const* d_in, const int* in_sizes, int n_in,
                              void* d_out, int out_size, void* d_ws,
                              size_t ws_size, hipStream_t stream) {
  (void)n_in; (void)out_size; (void)ws_size;
  const float* x      = (const float*)d_in[0];
  const float* ego_w0 = (const float*)d_in[1];
  const float* ego_b0 = (const float*)d_in[2];
  const float* ego_w1 = (const float*)d_in[3];
  const float* ego_b1 = (const float*)d_in[4];
  const float* oth_w0 = (const float*)d_in[5];
  const float* oth_b0 = (const float*)d_in[6];
  const float* oth_w1 = (const float*)d_in[7];
  const float* oth_b1 = (const float*)d_in[8];
  const float* Wk     = (const float*)d_in[9];
  const float* Wv     = (const float*)d_in[10];
  const float* Wq     = (const float*)d_in[11];
  const float* Wc     = (const float*)d_in[12];

  _Float16* wsw = (_Float16*)d_ws;   // 3*4096 halves = 24 KB, L2-resident

  // Prologue: swizzle GEMM weights into B-fragment order (6144 threads).
  stage_weights_kernel<<<dim3(24), dim3(256), 0, stream>>>(oth_w1, Wk, Wv, wsw);

  const int B = in_sizes[0] / (E_ * FIN);   // 8192
  ego_attention_kernel<<<dim3(B), dim3(BT), 0, stream>>>(
      x, ego_w0, ego_b0, ego_w1, ego_b1,
      oth_w0, oth_b0, oth_b1,
      Wq, Wc, wsw, (float*)d_out);
}
